// NSAAttention_50603304681857
// MI455X (gfx1250) — compile-verified, matrix operand order checked by reading, not run on hardware
//
#include <hip/hip_runtime.h>
#include <hip/hip_bf16.h>

// ---------------------------------------------------------------------------
// NSA attention forward for MI455X (gfx1250), wave32 + WMMA f16 (f32 accum).
// Projections and attention matmuls use v_wmma_f32_16x16x32_f16.
// GEMM stages B tiles into LDS with GLOBAL_LOAD_ASYNC_TO_LDS_B128
// (ASYNCcnt + s_wait_asynccnt), double-buffered, shared by 4 waves.
// ---------------------------------------------------------------------------

typedef __attribute__((ext_vector_type(16))) _Float16 v16h;
typedef __attribute__((ext_vector_type(8)))  _Float16 v8h;
typedef __attribute__((ext_vector_type(8)))  float    v8f;
typedef __attribute__((__vector_size__(4 * sizeof(int)))) int v4i_t;

union Frag {
  v16h v;
  v8h  h[2];
};

#define HID   1024
#define NH    16
#define HD    64
#define LSEQ  1024
#define BATCH 2
#define MROWS (BATCH * LSEQ)   // 2048
#define NBLK  32               // key blocks of 32
#define TOPK_ 8
#define WIN   128

#define AS1 __attribute__((address_space(1)))
#define AS3 __attribute__((address_space(3)))

__device__ __forceinline__ v8f wmma_f16(v16h a, v16h b, v8f c) {
  // (neg_a, A, neg_b, B, c_mod, C, reuse_a, reuse_b)
  return __builtin_amdgcn_wmma_f32_16x16x32_f16(false, a, false, b, (short)0, c,
                                                false, false);
}

__device__ __forceinline__ void wait_asynccnt0() {
#if __has_builtin(__builtin_amdgcn_s_wait_asynccnt)
  __builtin_amdgcn_s_wait_asynccnt(0);
#else
  asm volatile("s_wait_asynccnt 0x0" ::: "memory");
#endif
}

// Async copy of 16 bytes global -> LDS (per lane), tracked by ASYNCcnt.
__device__ __forceinline__ void async_load_b128(const _Float16* gsrc,
                                                _Float16* lds_dst) {
#if __has_builtin(__builtin_amdgcn_global_load_async_to_lds_b128)
  // Signature (from clang diagnostic): (v4i AS1*, v4i AS3*, imm offset, imm cpol)
  __builtin_amdgcn_global_load_async_to_lds_b128(
      (AS1 v4i_t*)gsrc, (AS3 v4i_t*)lds_dst, 0, 0);
#else
  // Generic LDS pointer = {shared_aperture_hi, lds_offset}; low 32 bits are
  // the LDS byte offset.
  unsigned loff = (unsigned)(uintptr_t)lds_dst;
  unsigned long long ga = (unsigned long long)(uintptr_t)gsrc;
  asm volatile("global_load_async_to_lds_b128 %0, %1, off"
               :: "v"(loff), "v"(ga)
               : "memory");
#endif
}

// ---------------------------------------------------------------------------
// Conversions
// ---------------------------------------------------------------------------
__global__ void cvt_f32_to_f16(const float* __restrict__ src,
                               _Float16* __restrict__ dst, int n) {
  int t = blockIdx.x * blockDim.x + threadIdx.x;
  if (t < n) dst[t] = (_Float16)src[t];
}

// W (K x N row-major, f32) -> Wt (N x K row-major, f16)
__global__ void cvt_weight_t(const float* __restrict__ W,
                             _Float16* __restrict__ Wt) {
  int t = blockIdx.x * blockDim.x + threadIdx.x;   // t < 1024*1024
  int n = t >> 10;
  int k = t & 1023;
  Wt[t] = (_Float16)W[(size_t)k * HID + n];
}

// ---------------------------------------------------------------------------
// WMMA GEMM: D = A(MxK,f16) * Bt^T + bias.
// Block = 128 threads (4 waves); each wave computes 32 rows x 64 cols.
// B tile (64 cols x 32 k, f16 = 4KB) is async-staged into LDS, double
// buffered, and shared by all 4 waves.  grid = (M/128, N/64).
// ---------------------------------------------------------------------------
__global__ __launch_bounds__(128)
void wmma_gemm(const _Float16* __restrict__ A,
               const _Float16* __restrict__ Bt,
               const float* __restrict__ bias,
               _Float16* __restrict__ outHead,
               _Float16* __restrict__ outHeadT,
               float* __restrict__ outF32,
               int K) {
  __shared__ _Float16 btile[2][64 * 32];

  const int tid = threadIdx.x;
  const int lane = tid & 31;
  const int w = tid >> 5;               // wave id 0..3
  const int g = lane >> 4;              // half-wave select
  const int ln = lane & 15;
  const int n0 = blockIdx.y * 64;       // col block
  const int rowb = blockIdx.x * 128 + w * 32;  // this wave's first row

  // issue the 4KB B tile for k-chunk kk into buffer buf (2 b128 per thread)
  auto issue = [&](int buf, int kk) {
#pragma unroll
    for (int r = 0; r < 2; ++r) {
      const int c = tid + r * 128;      // chunk 0..255 (16B each)
      const int n = c >> 2;             // 0..63
      const int kc = (c & 3) * 8;       // 0,8,16,24 halves
      async_load_b128(Bt + (size_t)(n0 + n) * K + kk + kc,
                      &btile[buf][n * 32 + kc]);
    }
  };

  v8f zero = {};
  v8f acc[2][4];
#pragma unroll
  for (int u = 0; u < 2; ++u)
#pragma unroll
    for (int t = 0; t < 4; ++t) acc[u][t] = zero;

  issue(0, 0);
  int pb = 0;
  for (int kk = 0; kk < K; kk += 32, pb ^= 1) {
    wait_asynccnt0();        // my async stores to LDS done
    __syncthreads();         // everyone's done -> tile pb valid
    if (kk + 32 < K) issue(pb ^ 1, kk + 32);

    const int kb = kk + g * 8;
    Frag a[2];
#pragma unroll
    for (int u = 0; u < 2; ++u) {
      const _Float16* ar = A + (size_t)(rowb + u * 16 + ln) * K;
      a[u].h[0] = *(const v8h*)(ar + kb);
      a[u].h[1] = *(const v8h*)(ar + kb + 16);
    }
#pragma unroll
    for (int t = 0; t < 4; ++t) {
      const _Float16* br = &btile[pb][(t * 16 + ln) * 32];
      Frag b;
      b.h[0] = *(const v8h*)(br + g * 8);
      b.h[1] = *(const v8h*)(br + g * 8 + 16);
#pragma unroll
      for (int u = 0; u < 2; ++u)
        acc[u][t] = wmma_f16(a[u].v, b.v, acc[u][t]);
    }
    __syncthreads();         // all reads of tile pb done before reuse
  }

#pragma unroll
  for (int u = 0; u < 2; ++u) {
#pragma unroll
    for (int t = 0; t < 4; ++t) {
#pragma unroll
      for (int e = 0; e < 8; ++e) {
        const int m = rowb + u * 16 + e + 8 * g;  // C/D row = e + 8*half
        const int n = n0 + t * 16 + ln;           // col = lane&15
        const float val = acc[u][t][e] + bias[n];
        if (outF32) outF32[(size_t)m * HID + n] = val;
        if (outHead || outHeadT) {
          const int b = m >> 10, l = m & 1023;
          const int h = n >> 6, d = n & 63;
          const int bh = b * NH + h;
          if (outHead)
            outHead[((size_t)bh * LSEQ + l) * HD + d] = (_Float16)val;
          if (outHeadT)
            outHeadT[((size_t)bh * HD + d) * LSEQ + l] = (_Float16)val;
        }
      }
    }
  }
}

// ---------------------------------------------------------------------------
// Compressed K/V: block means.  ck[(bh),blk,d]  cvt[(bh),d,blk]
// ---------------------------------------------------------------------------
__global__ void compress_kv(const _Float16* __restrict__ k,
                            const _Float16* __restrict__ vt,
                            _Float16* __restrict__ ck,
                            _Float16* __restrict__ cvt) {
  int t = blockIdx.x * blockDim.x + threadIdx.x;  // t < 32*32*64
  if (t >= BATCH * NH * NBLK * HD) return;
  const int d = t & 63;
  const int blk = (t >> 6) & 31;
  const int bh = t >> 11;
  float sk = 0.f, sv = 0.f;
  const _Float16* kp = k + ((size_t)bh * LSEQ + blk * 32) * HD + d;
  const _Float16* vp = vt + ((size_t)bh * HD + d) * LSEQ + blk * 32;
#pragma unroll
  for (int i = 0; i < 32; ++i) {
    sk += (float)kp[(size_t)i * HD];
    sv += (float)vp[i];
  }
  ck[((size_t)bh * NBLK + blk) * HD + d] = (_Float16)(sk * (1.f / 32.f));
  cvt[((size_t)bh * HD + d) * NBLK + blk] = (_Float16)(sv * (1.f / 32.f));
}

// ---------------------------------------------------------------------------
// Flash-attention kernel: one wave handles 16 query rows of one (b,h).
//   mode 0: compressed (keys = 32 block means; also accumulates block scores)
//   mode 1: selected   (keys = 8 top-k blocks of 32 contiguous rows)
//   mode 2: sliding    (causal window of 128, masked key tiles)
// grid = (L/16, B*NH), block = 32.
// ---------------------------------------------------------------------------
__global__ __launch_bounds__(32)
void attn_flash(const _Float16* __restrict__ q,
                const _Float16* __restrict__ kmat,
                const _Float16* __restrict__ vt,
                const int* __restrict__ selidx,
                float* __restrict__ out,
                float* __restrict__ bscore,
                int mode) {
  __shared__ _Float16 plds[16 * 32];

  const int lane = threadIdx.x;
  const int g = lane >> 4;
  const int ln = lane & 15;
  const int qt = blockIdx.x;
  const int bh = blockIdx.y;
  const int q0 = qt * 16;

  // Q A-fragments for both K-steps over head dim (64 = 2 x 32)
  const _Float16* qbase = q + ((size_t)bh * LSEQ + q0) * HD;
  Frag qa[2];
#pragma unroll
  for (int s = 0; s < 2; ++s) {
    const int kb = s * 32 + g * 8;
    qa[s].h[0] = *(const v8h*)(qbase + ln * HD + kb);
    qa[s].h[1] = *(const v8h*)(qbase + ln * HD + kb + 16);
  }

  v8f zero = {};
  v8f accO[4];
#pragma unroll
  for (int t = 0; t < 4; ++t) accO[t] = zero;
  float rowMax[8], rowSum[8];
#pragma unroll
  for (int e = 0; e < 8; ++e) { rowMax[e] = -1e30f; rowSum[e] = 0.f; }

  const int ldv = (mode == 0) ? NBLK : LSEQ;
  const _Float16* kbase = kmat + (size_t)bh * ((mode == 0) ? NBLK : LSEQ) * HD;
  const _Float16* vbase = vt + (size_t)bh * HD * ldv;

  int ktstart = 0, nTiles = 1;
  if (mode == 1) {
    nTiles = TOPK_;
  } else if (mode == 2) {
    int lo = q0 - (WIN - 1);
    if (lo < 0) lo = 0;
    ktstart = lo & ~31;
    nTiles = ((q0 + 15 - ktstart) >> 5) + 1;
  }

  for (int t = 0; t < nTiles; ++t) {
    int key0 = 0;
    if (mode == 1)      key0 = selidx[bh * TOPK_ + t] * 32;
    else if (mode == 2) key0 = ktstart + t * 32;

    // ---- scores: q (16x64) x K_tile^T (64x32) -> 16x32 f32 -------------
    v8f sc[2];
    sc[0] = zero; sc[1] = zero;
#pragma unroll
    for (int nt = 0; nt < 2; ++nt) {
      const _Float16* kr = kbase + (size_t)(key0 + nt * 16 + ln) * HD;
#pragma unroll
      for (int s = 0; s < 2; ++s) {
        const int kb = s * 32 + g * 8;
        Frag bf;
        bf.h[0] = *(const v8h*)(kr + kb);
        bf.h[1] = *(const v8h*)(kr + kb + 16);
        sc[nt] = wmma_f16(qa[s].v, bf.v, sc[nt]);
      }
    }

    // scale + (sliding) mask
#pragma unroll
    for (int nt = 0; nt < 2; ++nt) {
#pragma unroll
      for (int e = 0; e < 8; ++e) {
        float sv = sc[nt][e] * 0.125f;   // 1/sqrt(64)
        if (mode == 2) {
          const int i = q0 + e + 8 * g;
          const int j = key0 + nt * 16 + ln;
          if (!((j <= i) && (j > i - WIN))) sv = -1e9f;
        }
        sc[nt][e] = sv;
      }
    }

    // ---- online softmax (row spread over 16 lanes of each half) --------
    float tmax[8];
#pragma unroll
    for (int e = 0; e < 8; ++e) tmax[e] = fmaxf(sc[0][e], sc[1][e]);
    for (int w = 1; w < 16; w <<= 1) {
#pragma unroll
      for (int e = 0; e < 8; ++e)
        tmax[e] = fmaxf(tmax[e], __shfl_xor(tmax[e], w, 32));
    }
    float psum[8];
#pragma unroll
    for (int e = 0; e < 8; ++e) {
      const float nm = fmaxf(rowMax[e], tmax[e]);
      const float alpha = __expf(rowMax[e] - nm);
      rowMax[e] = nm;
      const float p0 = __expf(sc[0][e] - nm);
      const float p1 = __expf(sc[1][e] - nm);
      sc[0][e] = p0; sc[1][e] = p1;
      psum[e] = p0 + p1;
      rowSum[e] *= alpha;
#pragma unroll
      for (int t2 = 0; t2 < 4; ++t2) accO[t2][e] *= alpha;
    }
    for (int w = 1; w < 16; w <<= 1) {
#pragma unroll
      for (int e = 0; e < 8; ++e) psum[e] += __shfl_xor(psum[e], w, 32);
    }
#pragma unroll
    for (int e = 0; e < 8; ++e) rowSum[e] += psum[e];

    // compressed: single tile -> rowSum is final; accumulate block scores
    if (mode == 0) {
      float c0 = 0.f, c1 = 0.f;
#pragma unroll
      for (int e = 0; e < 8; ++e) {
        const float inv = 1.f / rowSum[e];
        c0 += sc[0][e] * inv;
        c1 += sc[1][e] * inv;
      }
      atomicAdd(bscore + bh * NBLK + ln, c0);
      atomicAdd(bscore + bh * NBLK + 16 + ln, c1);
    }

    // ---- C/D-layout P -> LDS -> A-fragment layout ----------------------
#pragma unroll
    for (int e = 0; e < 8; ++e) {
      const int m = e + 8 * g;
      plds[m * 32 + ln]      = (_Float16)sc[0][e];
      plds[m * 32 + 16 + ln] = (_Float16)sc[1][e];
    }
    __syncthreads();
    Frag pa;
    pa.h[0] = *(const v8h*)(&plds[ln * 32 + g * 8]);
    pa.h[1] = *(const v8h*)(&plds[ln * 32 + g * 8 + 16]);
    __syncthreads();

    // ---- PV: P (16x32) x V_tile (32x64), V loaded dim-major -----------
#pragma unroll
    for (int t2 = 0; t2 < 4; ++t2) {
      const _Float16* vr = vbase + (size_t)(t2 * 16 + ln) * ldv + key0;
      Frag vb;
      vb.h[0] = *(const v8h*)(vr + g * 8);
      vb.h[1] = *(const v8h*)(vr + g * 8 + 16);
      accO[t2] = wmma_f16(pa.v, vb.v, accO[t2]);
    }
  }

  // normalize and store f32
#pragma unroll
  for (int e = 0; e < 8; ++e) {
    const float inv = 1.f / rowSum[e];
    const size_t row = (size_t)bh * LSEQ + q0 + e + 8 * g;
#pragma unroll
    for (int t2 = 0; t2 < 4; ++t2)
      out[row * HD + t2 * 16 + ln] = accO[t2][e] * inv;
  }
}

// ---------------------------------------------------------------------------
// Top-k over 32 block scores per (b,h)
// ---------------------------------------------------------------------------
__global__ void topk_kernel(const float* __restrict__ bs, int* __restrict__ idx) {
  const int bh = threadIdx.x;
  if (bh >= BATCH * NH) return;
  float s[NBLK];
#pragma unroll
  for (int i = 0; i < NBLK; ++i) s[i] = bs[bh * NBLK + i];
  unsigned used = 0u;
  for (int t = 0; t < TOPK_; ++t) {
    float best = -1e30f;
    int barg = 0;
#pragma unroll
    for (int i = 0; i < NBLK; ++i) {
      const bool ok = ((used >> i) & 1u) == 0u;
      if (ok && s[i] > best) { best = s[i]; barg = i; }
    }
    used |= (1u << barg);
    idx[bh * TOPK_ + t] = barg;
  }
}

// ---------------------------------------------------------------------------
// Gates: sigmoid(hidden @ Wg + bg), tiny (H x 3)
// ---------------------------------------------------------------------------
__global__ void gates_kernel(const float* __restrict__ hs,
                             const float* __restrict__ Wg,
                             const float* __restrict__ bg,
                             float* __restrict__ gates) {
  const int m = blockIdx.x * blockDim.x + threadIdx.x;
  if (m >= MROWS) return;
  const float* x = hs + (size_t)m * HID;
  float a0 = bg[0], a1 = bg[1], a2 = bg[2];
  for (int i = 0; i < HID; ++i) {
    const float xi = x[i];
    a0 += xi * Wg[i * 3 + 0];
    a1 += xi * Wg[i * 3 + 1];
    a2 += xi * Wg[i * 3 + 2];
  }
  gates[m * 3 + 0] = 1.f / (1.f + __expf(-a0));
  gates[m * 3 + 1] = 1.f / (1.f + __expf(-a1));
  gates[m * 3 + 2] = 1.f / (1.f + __expf(-a2));
}

// ---------------------------------------------------------------------------
// Gated combine of the three attention outputs -> f16 activation for Wo GEMM
// ---------------------------------------------------------------------------
__global__ void combine_kernel(const float* __restrict__ oc,
                               const float* __restrict__ os,
                               const float* __restrict__ ow,
                               const float* __restrict__ gates,
                               _Float16* __restrict__ Xc) {
  const int t = blockIdx.x * blockDim.x + threadIdx.x;
  if (t >= MROWS * HID) return;
  const int m = t >> 10, n = t & 1023;
  const int b = m >> 10, l = m & 1023;
  const int h = n >> 6, d = n & 63;
  const size_t src = ((size_t)(b * NH + h) * LSEQ + l) * HD + d;
  const float* gm = gates + m * 3;
  Xc[t] = (_Float16)(gm[0] * oc[src] + gm[1] * os[src] + gm[2] * ow[src]);
}

// ---------------------------------------------------------------------------
// Host: workspace carving + launch sequence (graph-capture safe)
// ---------------------------------------------------------------------------
extern "C" void kernel_launch(void* const* d_in, const int* in_sizes, int n_in,
                              void* d_out, int out_size, void* d_ws, size_t ws_size,
                              hipStream_t stream) {
  (void)in_sizes; (void)n_in; (void)out_size; (void)ws_size;

  const float* hs = (const float*)d_in[0];
  const float* Wq = (const float*)d_in[1];
  const float* bq = (const float*)d_in[2];
  const float* Wk = (const float*)d_in[3];
  const float* bk = (const float*)d_in[4];
  const float* Wv = (const float*)d_in[5];
  const float* bv = (const float*)d_in[6];
  const float* Wg = (const float*)d_in[7];
  const float* bg = (const float*)d_in[8];
  const float* Wo = (const float*)d_in[9];
  const float* bo = (const float*)d_in[10];
  float* out = (float*)d_out;

  char* p = (char*)d_ws;
  auto alloc = [&](size_t bytes) -> void* {
    void* r = (void*)p;
    p += (bytes + 255) & ~(size_t)255;
    return r;
  };

  _Float16* Xh  = (_Float16*)alloc((size_t)MROWS * HID * 2);      // hidden f16
  _Float16* Wtq = (_Float16*)alloc((size_t)HID * HID * 2);
  _Float16* Wtk = (_Float16*)alloc((size_t)HID * HID * 2);
  _Float16* Wtv = (_Float16*)alloc((size_t)HID * HID * 2);
  _Float16* Wto = (_Float16*)alloc((size_t)HID * HID * 2);
  _Float16* qh  = (_Float16*)alloc((size_t)BATCH * NH * LSEQ * HD * 2);
  _Float16* kh  = (_Float16*)alloc((size_t)BATCH * NH * LSEQ * HD * 2);
  _Float16* vth = (_Float16*)alloc((size_t)BATCH * NH * HD * LSEQ * 2);
  _Float16* ck  = (_Float16*)alloc((size_t)BATCH * NH * NBLK * HD * 2);
  _Float16* cvt = (_Float16*)alloc((size_t)BATCH * NH * HD * NBLK * 2);
  float* bscore = (float*)alloc((size_t)BATCH * NH * NBLK * 4);
  int*   selidx = (int*)alloc((size_t)BATCH * NH * TOPK_ * 4);
  float* oc     = (float*)alloc((size_t)BATCH * NH * LSEQ * HD * 4);
  float* osel   = (float*)alloc((size_t)BATCH * NH * LSEQ * HD * 4);
  float* ow     = (float*)alloc((size_t)BATCH * NH * LSEQ * HD * 4);
  float* gates  = (float*)alloc((size_t)MROWS * 3 * 4);
  _Float16* Xc  = (_Float16*)alloc((size_t)MROWS * HID * 2);

  const int nHid = MROWS * HID;       // 2,097,152
  const int nW   = HID * HID;         // 1,048,576

  // 1) conversions
  cvt_f32_to_f16<<<(nHid + 255) / 256, 256, 0, stream>>>(hs, Xh, nHid);
  cvt_weight_t<<<(nW + 255) / 256, 256, 0, stream>>>(Wq, Wtq);
  cvt_weight_t<<<(nW + 255) / 256, 256, 0, stream>>>(Wk, Wtk);
  cvt_weight_t<<<(nW + 255) / 256, 256, 0, stream>>>(Wv, Wtv);
  cvt_weight_t<<<(nW + 255) / 256, 256, 0, stream>>>(Wo, Wto);

  // 2) Q/K/V projections (WMMA + async-LDS B staging)
  dim3 ggrid(MROWS / 128, HID / 64);
  wmma_gemm<<<ggrid, 128, 0, stream>>>(Xh, Wtq, bq, qh, nullptr, nullptr, HID);
  wmma_gemm<<<ggrid, 128, 0, stream>>>(Xh, Wtk, bk, kh, nullptr, nullptr, HID);
  wmma_gemm<<<ggrid, 128, 0, stream>>>(Xh, Wtv, bv, nullptr, vth, nullptr, HID);

  // 3) compressed K/V block means
  const int nC = BATCH * NH * NBLK * HD;  // 65536
  compress_kv<<<(nC + 255) / 256, 256, 0, stream>>>(kh, vth, ck, cvt);

  // 4) compressed attention (+ block scores)
  (void)hipMemsetAsync(bscore, 0, (size_t)BATCH * NH * NBLK * 4, stream);
  dim3 agrid(LSEQ / 16, BATCH * NH);
  attn_flash<<<agrid, 32, 0, stream>>>(qh, ck, cvt, nullptr, oc, bscore, 0);

  // 5) top-k block selection
  topk_kernel<<<1, 32, 0, stream>>>(bscore, selidx);

  // 6) selected attention
  attn_flash<<<agrid, 32, 0, stream>>>(qh, kh, vth, selidx, osel, nullptr, 1);

  // 7) sliding-window attention
  attn_flash<<<agrid, 32, 0, stream>>>(qh, kh, vth, nullptr, ow, nullptr, 2);

  // 8) gates + gated combine
  gates_kernel<<<(MROWS + 255) / 256, 256, 0, stream>>>(hs, Wg, bg, gates);
  combine_kernel<<<(nHid + 255) / 256, 256, 0, stream>>>(oc, osel, ow, gates, Xc);

  // 9) output projection (WMMA) -> d_out f32
  wmma_gemm<<<ggrid, 128, 0, stream>>>(Xc, Wto, bo, nullptr, nullptr, out, HID);
}
